// ChebyshevLayer_33629593928064
// MI455X (gfx1250) — matrix operand order repaired and checked by reading, not compiled, not run on hardware
//
#include <hip/hip_runtime.h>

// ---------------------------------------------------------------------------
// ChebyshevLayer on MI455X (gfx1250):
//   - big DCT GEMMs via V_WMMA_F32_16X16X32_F16, Markidis hi/lo fp16 split
//     (Ah*Bh + Ah*Bl + Al*Bh, f32 accum) -> ~fp32 accuracy at 16-bit XDL rate
//   - B operand pre-tiled in GLOBAL memory in the exact LDS fragment image,
//     staged with GLOBAL_LOAD_ASYNC_TO_LDS_B64 (ASYNCcnt)
//   - per-mode 64x64x64 channel mixing via exact V_WMMA_F32_16X16X4_F32
// ---------------------------------------------------------------------------

typedef float    v2f  __attribute__((ext_vector_type(2)));
typedef float    v8f  __attribute__((ext_vector_type(8)));
typedef _Float16 v8h  __attribute__((ext_vector_type(8)));
typedef _Float16 v16h __attribute__((ext_vector_type(16)));
typedef int      v2i  __attribute__((ext_vector_type(2)));

#define WMMA_F32(a, b, c) \
    __builtin_amdgcn_wmma_f32_16x16x4_f32(false, (a), false, (b), (short)0, (c), false, false)
#define WMMA_F16(a, b, c) \
    __builtin_amdgcn_wmma_f32_16x16x32_f16(false, (a), false, (b), (short)0, (c), false, false)

#if defined(__has_builtin)
#  if __has_builtin(__builtin_amdgcn_global_load_async_to_lds_b64)
#    define HAVE_ASYNC_LDS 1
#  endif
#  if __has_builtin(__builtin_amdgcn_s_wait_asynccnt)
#    define HAVE_WAIT_ASYNC 1
#  endif
#endif
#ifndef HAVE_ASYNC_LDS
#  define HAVE_ASYNC_LDS 0
#endif
#ifndef HAVE_WAIT_ASYNC
#  define HAVE_WAIT_ASYNC 0
#endif

static constexpr int Bsz   = 64;
static constexpr int NGRID = 2048;
static constexpr int MODES = 512;
static constexpr int BI    = Bsz * 64;          // 4096 rows for GEMM1
static constexpr int BO    = Bsz * 64;          // 4096 rows for GEMM3
static constexpr int AP    = 24;                // fragment slot stride (halves)
static constexpr int TILEH = 64 * 2 * AP;       // halves per B tile (64 cols x 32 K)
static constexpr long TB   = 512L * TILEH;      // halves per tiled B table (512 tiles)

// ---- async copy of one 8-byte chunk: global -> LDS (fallback: synchronous)
__device__ inline void async_cp_b64(void* lds_dst, const void* g_src) {
#if HAVE_ASYNC_LDS
    void* gs = const_cast<void*>(g_src);
    __builtin_amdgcn_global_load_async_to_lds_b64(
        (__attribute__((address_space(1))) v2i*)gs,
        (__attribute__((address_space(3))) v2i*)lds_dst, 0, 0);
#else
    *(unsigned long long*)lds_dst = *(const unsigned long long*)g_src;
#endif
}
__device__ inline void async_wait0() {
#if HAVE_ASYNC_LDS
#  if HAVE_WAIT_ASYNC
    __builtin_amdgcn_s_wait_asynccnt(0);
#  else
    asm volatile("s_wait_asynccnt 0" ::: "memory");
#  endif
#endif
}

// split f32 -> f16 hi/lo
__device__ inline void split_hl(float v, _Float16& h, _Float16& l) {
    h = (_Float16)v;
    l = (_Float16)(v - (float)h);
}

// load 16 contiguous halves from LDS as a v16h (two 16B-aligned v8h loads)
__device__ inline v16h ld_frag(const _Float16* p) {
    union { v16h v; v8h h[2]; } u;
    u.h[0] = *(const v8h*)(p);
    u.h[1] = *(const v8h*)(p + 8);
    return u.v;
}

// ---------------------------------------------------------------------------
// Build DCT-I matrices, hi/lo split, directly in the tiled LDS fragment image:
//   tile t = tc*KCH + kc covers cols [tc*64,+64), K [kc*32,+32)
//   element (K=kk, col=nn) -> t*TILEH + (nn*2 + (kk>>4))*AP + (kk&15)
// B1 (forward): K-dim = grid n (KCH=64), col = mode k; value cos*c_n
// B3 (inverse): K-dim = mode k (KCH=16), col = grid n; value cos*c_k
// Angle arithmetic replicates jnp fp32: f32(pi)*f32(k*n)/2047 (k*n<2^24 exact).
// ---------------------------------------------------------------------------
__global__ __launch_bounds__(256) void build_dct(_Float16* __restrict__ B1h,
                                                 _Float16* __restrict__ B1l,
                                                 _Float16* __restrict__ B3h,
                                                 _Float16* __restrict__ B3l) {
    const long idx = (long)blockIdx.x * 256 + threadIdx.x;   // over 2048*512
    const int n = (int)(idx >> 9);     // grid point 0..2047
    const int k = (int)(idx & 511);    // mode       0..511
    float p   = (float)(n * k);
    float ang = 3.14159274101257324219f * p;
    ang       = ang / 2047.0f;
    const float cp = cosf(ang);
    const float cn = (n == 0 || n == NGRID - 1) ? 1.0f : 2.0f;
    const float ck = (k == 0) ? 1.0f : 2.0f;   // k < 512 < 2047

    _Float16 h, l;
    // B1: tc = k/64, kc = n/32, col nn = k%64, K kk = n%32
    {
        const long t   = (long)(k >> 6) * (NGRID / 32) + (n >> 5);
        const int  kk  = n & 31;
        const long off = t * TILEH + (long)((k & 63) * 2 + (kk >> 4)) * AP + (kk & 15);
        split_hl(cp * cn, h, l);
        B1h[off] = h; B1l[off] = l;
    }
    // B3: tc = n/64, kc = k/32, col nn = n%64, K kk = k%32
    {
        const long t   = (long)(n >> 6) * (MODES / 32) + (k >> 5);
        const int  kk  = k & 31;
        const long off = t * TILEH + (long)((n & 63) * 2 + (kk >> 4)) * AP + (kk & 15);
        split_hl(cp * ck, h, l);
        B3h[off] = h; B3l[off] = l;
    }
}

// ---------------------------------------------------------------------------
// Split-fp16 WMMA GEMM: C[r,c] = sum_k A[r,k] * B[k,c]
//   A f32 at A[r*Ars + k*Aks]; ACOL selects coalesced LDS-fill mapping.
//   B pre-tiled hi/lo half tables (fragment image) -> raw async copy to LDS.
//   C f32 at C[r*Crs + c*Ccs].
// Block 256 thr (8 waves), BM=128, BN=64, KC=32; wave owns 32x32 (2x2 tiles).
// Fragment maps (16-bit, contiguous k order used here):
//   A (16x32): K = j + g*16   B (32x16): K = j + g*16
// ---------------------------------------------------------------------------
template <bool ACOL>
__global__ __launch_bounds__(256) void gemm_wmma_f16split(
    const float* __restrict__ A, long Ars, long Aks,
    const _Float16* __restrict__ Bht, const _Float16* __restrict__ Blt,
    float* __restrict__ C, long Crs, long Ccs, int K) {
    constexpr int BM = 128, BN = 64, KC = 32;
    __shared__ __align__(16) _Float16 AhS[BM * 2][AP];
    __shared__ __align__(16) _Float16 AlS[BM * 2][AP];
    __shared__ __align__(16) _Float16 BhS[BN * 2 * AP];
    __shared__ __align__(16) _Float16 BlS[BN * 2 * AP];

    const int tid   = threadIdx.x;
    const int lane  = tid & 31;
    const int wave  = tid >> 5;
    const int wm    = wave >> 1;          // 0..3
    const int wn    = wave & 1;           // 0..1
    const int laneM = lane & 15;
    const int g     = lane >> 4;          // lane group (K-half select)
    const int rGrp  = g * 8;              // C/D row group offset

    const long rowBlock = (long)blockIdx.y * BM;
    const long colBlock = (long)blockIdx.x * BN;
    const int  KCH      = K >> 5;         // K chunks (tiles along K)

    v8f acc[2][2];
    #pragma unroll
    for (int i = 0; i < 2; ++i)
        #pragma unroll
        for (int j = 0; j < 2; ++j)
            #pragma unroll
            for (int r = 0; r < 8; ++r) acc[i][j][r] = 0.0f;

    for (int k0 = 0; k0 < K; k0 += KC) {
        // ---- B tiles: raw async copy, fragment image already built in global
        {
            const long t = (long)blockIdx.x * KCH + (k0 >> 5);
            const _Float16* srcH = Bht + t * TILEH;
            const _Float16* srcL = Blt + t * TILEH;
            #pragma unroll
            for (int c = 0; c < 3; ++c) {                 // 768 8B-chunks / 256 thr
                const int off = (tid + c * 256) * 4;      // halves
                async_cp_b64(&BhS[off], srcH + off);
                async_cp_b64(&BlS[off], srcL + off);
            }
        }
        // ---- A tile (128x32 f32 -> split hi/lo, fragment-order scatter);
        //      global load latency overlaps the async B copies above
        #pragma unroll
        for (int i = 0; i < (BM * KC) / 256; ++i) {
            const int e = tid + i * 256;
            int r, k;
            if (!ACOL) { r = e >> 5;  k = e & 31; }   // k-fast: A row-major
            else       { r = e & 127; k = e >> 7; }   // r-fast: A col-major
            const float v = A[(rowBlock + r) * Ars + (long)(k0 + k) * Aks];
            _Float16 h, l;
            split_hl(v, h, l);
            int gg, j;
            gg = k >> 4; j = k & 15;
            AhS[r * 2 + gg][j] = h;
            AlS[r * 2 + gg][j] = l;
        }
        async_wait0();        // this wave's async B chunks are in LDS
        __syncthreads();      // everyone's chunks visible

        // ---- fragments: 2 row tiles, 2 col tiles, hi+lo each
        const int r0 = wm * 32 + laneM;
        const int c0 = wn * 32 + laneM;
        v16h ah[2], al[2], bh[2], bl[2];
        ah[0] = ld_frag(&AhS[r0 * 2 + g][0]);
        al[0] = ld_frag(&AlS[r0 * 2 + g][0]);
        ah[1] = ld_frag(&AhS[(r0 + 16) * 2 + g][0]);
        al[1] = ld_frag(&AlS[(r0 + 16) * 2 + g][0]);
        bh[0] = ld_frag(&BhS[(c0 * 2 + g) * AP]);
        bl[0] = ld_frag(&BlS[(c0 * 2 + g) * AP]);
        bh[1] = ld_frag(&BhS[((c0 + 16) * 2 + g) * AP]);
        bl[1] = ld_frag(&BlS[((c0 + 16) * 2 + g) * AP]);

        // ---- 12 WMMA: hi*hi + hi*lo + lo*hi per tile
        #pragma unroll
        for (int im = 0; im < 2; ++im)
            #pragma unroll
            for (int jn = 0; jn < 2; ++jn) {
                acc[im][jn] = WMMA_F16(ah[im], bh[jn], acc[im][jn]);
                acc[im][jn] = WMMA_F16(ah[im], bl[jn], acc[im][jn]);
                acc[im][jn] = WMMA_F16(al[im], bh[jn], acc[im][jn]);
            }
        __syncthreads();
    }

    // ---- store D: VGPR r -> row r (lanes 0-15) / r+8 (lanes 16-31)
    #pragma unroll
    for (int im = 0; im < 2; ++im)
        #pragma unroll
        for (int jn = 0; jn < 2; ++jn)
            #pragma unroll
            for (int r = 0; r < 8; ++r) {
                const long row = rowBlock + wm * 32 + im * 16 + r + rGrp;
                const long col = colBlock + wn * 32 + jn * 16 + laneM;
                C[row * Crs + col * Ccs] = acc[im][jn][r];
            }
}

// ---------------------------------------------------------------------------
// Per-mode channel mixing (exact f32 WMMA), one workgroup per mode:
//   om[m][b*64+o] = sum_i xc[m][b*64+i] * w[(i*64+o)*512 + m]
// ---------------------------------------------------------------------------
__global__ __launch_bounds__(256) void mode_mix(const float* __restrict__ xc,
                                                const float* __restrict__ w,
                                                float* __restrict__ om) {
    __shared__ float As[64][66];   // A[b][i]
    __shared__ float Bs[64][66];   // B[i][o]
    const int m    = blockIdx.x;
    const int tid  = threadIdx.x;
    const int lane = tid & 31;
    const int wave = tid >> 5;

    #pragma unroll
    for (int i = 0; i < 16; ++i) {
        const int e = tid + i * 256;                   // 0..4095
        As[e >> 6][e & 63] = xc[(long)m * BI + e];     // contiguous -> coalesced
        Bs[e >> 6][e & 63] = w[(long)e * MODES + m];   // strided gather (L2 hit)
    }
    __syncthreads();

    const int rt    = wave >> 1;
    const int ct0   = (wave & 1) * 2;
    const int laneM = lane & 15;
    const int kb    = (lane >> 4) * 2;
    const int rGrp  = (lane >> 4) * 8;

    v8f acc0, acc1;
    #pragma unroll
    for (int r = 0; r < 8; ++r) { acc0[r] = 0.0f; acc1[r] = 0.0f; }

    #pragma unroll
    for (int kk = 0; kk < 64; kk += 4) {
        v2f a, b0, b1;
        a.x  = As[rt * 16 + laneM][kk + kb];
        a.y  = As[rt * 16 + laneM][kk + kb + 1];
        b0.x = Bs[kk + kb][ct0 * 16 + laneM];
        b0.y = Bs[kk + kb + 1][ct0 * 16 + laneM];
        b1.x = Bs[kk + kb][ct0 * 16 + 16 + laneM];
        b1.y = Bs[kk + kb + 1][ct0 * 16 + 16 + laneM];
        acc0 = WMMA_F32(a, b0, acc0);
        acc1 = WMMA_F32(a, b1, acc1);
    }

    #pragma unroll
    for (int r = 0; r < 8; ++r) {
        const int row = rt * 16 + r + rGrp;
        om[(long)m * BO + row * 64 + (ct0 * 16 + laneM)]      = acc0[r];
        om[(long)m * BO + row * 64 + (ct0 * 16 + 16 + laneM)] = acc1[r];
    }
}

// ---------------------------------------------------------------------------
// Launch. Workspace: B1h/B1l/B3h/B3l tiled tables 4x3MB | xc 8MB | om 8MB = 28MB.
// ---------------------------------------------------------------------------
extern "C" void kernel_launch(void* const* d_in, const int* in_sizes, int n_in,
                              void* d_out, int out_size, void* d_ws, size_t ws_size,
                              hipStream_t stream) {
    const float* x = (const float*)d_in[0];   // [64, 64, 2048]
    const float* w = (const float*)d_in[1];   // [64, 64, 512]
    float* out = (float*)d_out;               // [64, 64, 2048]

    _Float16* B1h = (_Float16*)d_ws;
    _Float16* B1l = B1h + TB;
    _Float16* B3h = B1l + TB;
    _Float16* B3l = B3h + TB;
    float*    xc  = (float*)(B3l + TB);       // [512][4096] mode-major
    float*    om  = xc + (long)MODES * BI;    // [512][4096] mode-major

    // 1) pre-split, pre-tiled DCT-I matrices (4096 blocks exactly)
    build_dct<<<(NGRID * MODES) / 256, 256, 0, stream>>>(B1h, B1l, B3h, B3l);

    // 2) x_cheb = x (4096x2048 row-major) @ B1 (2048x512) -> xc[mode][bi]
    gemm_wmma_f16split<false><<<dim3(MODES / 64, BI / 128), 256, 0, stream>>>(
        x, /*Ars=*/NGRID, /*Aks=*/1, B1h, B1l,
        xc, /*Crs=*/1, /*Ccs=*/BI, /*K=*/NGRID);

    // 3) per-mode 64x64x64 channel mixing (exact f32 WMMA)
    mode_mix<<<MODES, 256, 0, stream>>>(xc, w, om);

    // 4) out = om (4096x512 via mode-major) @ B3 (512x2048) -> d_out
    gemm_wmma_f16split<true><<<dim3(NGRID / 64, BO / 128), 256, 0, stream>>>(
        om, /*Ars=*/1, /*Aks=*/BO, B3h, B3l,
        out, /*Crs=*/NGRID, /*Ccs=*/1, /*K=*/MODES);
}
